// FLoRALayer_44143673868402
// MI455X (gfx1250) — compile-verified, matrix operand order checked by reading, not run on hardware
//
#include <hip/hip_runtime.h>

typedef __attribute__((ext_vector_type(16))) _Float16 v16h;
typedef __attribute__((ext_vector_type(8)))  float    v8f;
typedef __attribute__((ext_vector_type(4)))  float    f32x4;
typedef __attribute__((ext_vector_type(4)))  unsigned u32x4;

#define B_  8
#define S_  2048
#define D_  1024
#define R_  8

#define BM  256
#define BN  128
#define BK  32
#define NK  (D_ / BK)   // 32 k-steps

// Load a 16-half WMMA fragment slice from LDS: halves [0..7] and [16..23]
// relative to `p` (which already includes the lane's koff of 0 or 8).
__device__ __forceinline__ v16h load_frag16(const _Float16* p) {
    union { u32x4 u[2]; v16h h; } t;
    const u32x4* q = (const u32x4*)p;
    t.u[0] = q[0];   // halves k+0..k+7   (16B aligned)
    t.u[1] = q[2];   // halves k+16..k+23 (+32 bytes)
    return t.h;
}

__global__ __launch_bounds__(256)
void flora_wmma_kernel(const float* __restrict__ x,
                       const float* __restrict__ adapter_b,
                       const float* __restrict__ adapter_a,
                       const float* __restrict__ W0,
                       float* __restrict__ out)
{
    // double-buffered tiles: A = x (f16), B = (adapter_b-scaled W0)^T (f16)
    __shared__ __align__(16) _Float16 As[2][BM * BK];  // 2 x 16 KB
    __shared__ __align__(16) _Float16 Bs[2][BN * BK];  // 2 x  8 KB

    const int tid  = threadIdx.x;
    const int lane = tid & 31;
    const int wave = tid >> 5;
    const int wr   = wave & 3;          // 4 row-groups of 64 rows
    const int wc   = wave >> 2;         // 2 col-groups of 64 cols

    const int s0 = blockIdx.x * BM;     // S tile
    const int n0 = blockIdx.y * BN;     // O tile
    const int bz = blockIdx.z;          // (b, r)
    const int b  = bz >> 3;
    const int r  = bz & 7;

    const int m0w    = wr * 64;
    const int n0w    = wc * 64;
    const int j2base = (n0 >> 3) + wc * 8;   // output-column base for this wave

    // Per-lane adapter_a weights, one per 16-wide column fragment.
    // Reduction group = 8 adjacent lanes; lane&7 = r2, (lane>>3)&1 = which j2.
    float wv[4];
    {
        const int r2  = lane & 7;
        const int par = (lane >> 3) & 1;
#pragma unroll
        for (int c = 0; c < 4; ++c) {
            const int j2 = j2base + c * 2 + par;
            wv[c] = adapter_a[((size_t)b * R_ + r2) * D_ + r * 128 + j2];
        }
    }

    v8f acc[4][4];
#pragma unroll
    for (int i = 0; i < 4; ++i)
#pragma unroll
        for (int j = 0; j < 4; ++j)
            acc[i][j] = (v8f){0.f, 0.f, 0.f, 0.f, 0.f, 0.f, 0.f, 0.f};

    // ---- loader thread mapping (uniform across the block) ----
    // A: each thread owns one full row (32 k's = 8 float4 loads, 4 b128 stores)
    // B: each thread owns two k-rows x 8 o's (transpose + scale)
    const int kp = (tid >> 4) * 2;       // 0,2,...,30
    const int og = (tid & 15) * 8;       // 0..120

    const float* xrow = x + ((size_t)b * S_ + (s0 + tid)) * D_;
    const float* bcol = adapter_b + (size_t)b * D_ * R_ + r;   // stride R over d

    f32x4 xa[8];
    f32x4 wb[4];
    float sc0, sc1;

    auto gload = [&](int k0) {
        const float* xp = xrow + k0;
#pragma unroll
        for (int i = 0; i < 8; ++i)
            xa[i] = ((const f32x4*)xp)[i];

        const float* w0p = W0 + (size_t)(k0 + kp) * D_ + n0 + og;
        wb[0] = ((const f32x4*)w0p)[0];
        wb[1] = ((const f32x4*)w0p)[1];
        wb[2] = ((const f32x4*)(w0p + D_))[0];
        wb[3] = ((const f32x4*)(w0p + D_))[1];
        sc0 = bcol[(size_t)(k0 + kp) * R_];
        sc1 = bcol[(size_t)(k0 + kp + 1) * R_];

        if (k0 + BK < D_) {
            __builtin_prefetch(xp + BK, 0, 1);                 // global_prefetch_b8
            __builtin_prefetch(w0p + (size_t)BK * D_, 0, 1);
        }
    };

    auto sstore = [&](int p) {
        {   // x row: 32 f32 -> f16, four ds_store_b128
            union { _Float16 h[32]; u32x4 u[4]; } pk;
#pragma unroll
            for (int i = 0; i < 8; ++i) {
                pk.h[4 * i + 0] = (_Float16)xa[i].x;
                pk.h[4 * i + 1] = (_Float16)xa[i].y;
                pk.h[4 * i + 2] = (_Float16)xa[i].z;
                pk.h[4 * i + 3] = (_Float16)xa[i].w;
            }
            u32x4* dst = (u32x4*)&As[p][tid * BK];
#pragma unroll
            for (int i = 0; i < 4; ++i)
                dst[i] = pk.u[i];
        }
        {   // W0 tile: scale rows by adapter_b, transpose into [o][k]
            float r0[8] = {wb[0].x, wb[0].y, wb[0].z, wb[0].w,
                           wb[1].x, wb[1].y, wb[1].z, wb[1].w};
            float r1[8] = {wb[2].x, wb[2].y, wb[2].z, wb[2].w,
                           wb[3].x, wb[3].y, wb[3].z, wb[3].w};
#pragma unroll
            for (int oo = 0; oo < 8; ++oo) {
                union { _Float16 h[2]; unsigned u; } p2;
                p2.h[0] = (_Float16)(r0[oo] * sc0);
                p2.h[1] = (_Float16)(r1[oo] * sc1);
                *(unsigned*)&Bs[p][(og + oo) * BK + kp] = p2.u;  // ds_store_b32
            }
        }
    };

    const int lrow = lane & 15;
    const int koff = (lane < 16) ? 0 : 8;

    auto compute = [&](int p) {
        v16h bf[4];
#pragma unroll
        for (int c = 0; c < 4; ++c)
            bf[c] = load_frag16(&Bs[p][(n0w + c * 16 + lrow) * BK + koff]);

#pragma unroll
        for (int rt = 0; rt < 4; ++rt) {
            v16h af = load_frag16(&As[p][(m0w + rt * 16 + lrow) * BK + koff]);
#pragma unroll
            for (int c = 0; c < 4; ++c)
                acc[rt][c] = __builtin_amdgcn_wmma_f32_16x16x32_f16(
                    false, af, false, bf[c], (short)0, acc[rt][c],
                    false, false);
        }
    };

    // ---------- prologue: fill buffer 0
    gload(0);
    sstore(0);
    __syncthreads();

    // ---------- main loop: one barrier per k-step (double-buffered)
    for (int kt = 0; kt < NK; ++kt) {
        const int p = kt & 1;
        if (kt + 1 < NK) gload((kt + 1) * BK);   // long-latency first
        compute(p);                              // 16 WMMAs from buffer p
        if (kt + 1 < NK) sstore(p ^ 1);          // convert + stage next
        __syncthreads();
    }

    // ---------- epilogue: weighted butterfly over 8-lane groups, relu(mean)
    const bool storeLane = ((lane & 7) == 0);
    const int  rowAdd    = (lane & 16) ? 8 : 0;
    const int  par       = (lane >> 3) & 1;

#pragma unroll
    for (int rt = 0; rt < 4; ++rt) {
#pragma unroll
        for (int c = 0; c < 4; ++c) {
            const float w = wv[c];
#pragma unroll
            for (int v = 0; v < 8; ++v) {
                float val = acc[rt][c][v] * w;
                val += __int_as_float(__builtin_amdgcn_ds_swizzle(__float_as_int(val), 0x041F)); // SWAPX1
                val += __int_as_float(__builtin_amdgcn_ds_swizzle(__float_as_int(val), 0x081F)); // SWAPX2
                val += __int_as_float(__builtin_amdgcn_ds_swizzle(__float_as_int(val), 0x101F)); // SWAPX4
                if (storeLane) {
                    const int row = s0 + m0w + rt * 16 + v + rowAdd;
                    const int d2  = r * 128 + j2base + c * 2 + par;
                    out[((size_t)b * S_ + row) * D_ + d2] = fmaxf(val * 0.125f, 0.f);
                }
            }
        }
    }
}

extern "C" void kernel_launch(void* const* d_in, const int* in_sizes, int n_in,
                              void* d_out, int out_size, void* d_ws, size_t ws_size,
                              hipStream_t stream) {
    const float* x         = (const float*)d_in[0];
    const float* adapter_b = (const float*)d_in[1];
    const float* adapter_a = (const float*)d_in[2];
    const float* W0        = (const float*)d_in[3];
    float* out = (float*)d_out;

    dim3 grid(S_ / BM, D_ / BN, B_ * R_);   // (8, 8, 64) = 4096 workgroups
    dim3 block(256);                        // 8 wave32 waves, 4x2 wave grid
    flora_wmma_kernel<<<grid, block, 0, stream>>>(x, adapter_b, adapter_a, W0, out);
}
